// MinVQVAE1D_82102594830938
// MI455X (gfx1250) — compile-verified, matrix operand order checked by reading, not run on hardware
//
#include <hip/hip_runtime.h>
#include <hip/hip_bf16.h>
#include <stdint.h>

typedef __bf16 bf16;
typedef __attribute__((ext_vector_type(16))) __bf16 v16bf;
typedef __attribute__((ext_vector_type(8)))  float  v8f;

// ---------------------------------------------------------------------------
// TDM availability (device pass only). Fallback = global_load -> ds_store.
// ---------------------------------------------------------------------------
#if defined(__AMDGCN__) && defined(__has_builtin)
#  if __has_builtin(__builtin_amdgcn_tensor_load_to_lds)
#    define USE_TDM 1
#  endif
#endif
#ifndef USE_TDM
#  define USE_TDM 0
#endif

#if USE_TDM
typedef __attribute__((ext_vector_type(4))) unsigned int tdm_v4u;
typedef __attribute__((ext_vector_type(8))) int          tdm_v8i;
typedef __attribute__((ext_vector_type(4))) int          tdm_v4i;

__device__ __forceinline__ unsigned lds_off(const void* p) {
  // Flat LDS address: low 32 bits are the LDS offset (aperture in high bits).
  return (unsigned)(unsigned long long)(uintptr_t)p;
}

// 2D tile load: tileRows x tileCols (elements, 2B each), row stride = strideElems.
__device__ __forceinline__ void tdm_load_2d(const void* gptr, unsigned ldsAddr,
                                            int strideElems, int tileRows, int tileCols)
{
  unsigned long long ga = (unsigned long long)(uintptr_t)gptr;
  tdm_v4u g0;
  g0.x = 1u;                                                    // count=1, user D#
  g0.y = ldsAddr;                                               // lds_addr (bytes)
  g0.z = (unsigned)ga;                                          // global_addr[31:0]
  g0.w = (unsigned)((ga >> 32) & 0x01FFFFFFu) | (2u << 30);     // global_addr[56:32] | type=2
  tdm_v8i g1;
  g1[0] = (int)(1u << 16);                                      // data_size=2B; no mask/pad/iter
  g1[1] = (int)(((unsigned)strideElems & 0xFFFFu) << 16);       // tensor_dim0[15:0]
  g1[2] = (int)((((unsigned)strideElems >> 16) & 0xFFFFu)
                | (0xFFFFu << 16));                             // tensor_dim0[31:16] | tensor_dim1 lo (big)
  g1[3] = (int)((unsigned)tileCols << 16);                      // tensor_dim1 hi=0 | tile_dim0
  g1[4] = (int)(unsigned)tileRows;                              // tile_dim1 | tile_dim2=0
  g1[5] = (int)strideElems;                                     // tensor_dim0_stride[31:0]
  g1[6] = 0;                                                    // stride[47:32] | dim1_stride lo
  g1[7] = 0;
  tdm_v4i gz = {0, 0, 0, 0};
#if __clang_major__ >= 23
  tdm_v8i gz8 = {0, 0, 0, 0, 0, 0, 0, 0};
  __builtin_amdgcn_tensor_load_to_lds(g0, g1, gz, gz, gz8, 0);
#else
  __builtin_amdgcn_tensor_load_to_lds(g0, g1, gz, gz, 0);
#endif
}
#endif // USE_TDM

// ---------------------------------------------------------------------------
// WMMA helpers (bf16 -> f32, 16x16x32)
// ---------------------------------------------------------------------------
union FragU { uint4 q[2]; v16bf v; };

// Fragment layout (16-bit A/B, wave32): frag[0..7] = K 8h..8h+7, frag[8..15] = K 16+8h..
// s points at the start of this lane's 32-element K row in LDS; h = lane>>4.
__device__ __forceinline__ v16bf ld_frag(const bf16* s, int h) {
  FragU f;
  f.q[0] = *(const uint4*)(s + 8 * h);
  f.q[1] = *(const uint4*)(s + 16 + 8 * h);
  return f.v;
}

__device__ __forceinline__ v8f wmma_bf16(v16bf a, v16bf b, v8f c) {
  return __builtin_amdgcn_wmma_f32_16x16x32_bf16(false, a, false, b, (short)0, c, false, false);
}

__device__ __forceinline__ float gelu_exact(float v) {
  return 0.5f * v * (1.0f + erff(v * 0.70710678118654752f));
}
__device__ __forceinline__ float sigmoid_(float v) {
  return 1.0f / (1.0f + __expf(-v));
}

// ---------------------------------------------------------------------------
// Tiled GEMM: C[M,N] = act(A[M,K] @ W + bias), W given pre-transposed as Bt[N,K].
// Block = 128x128 tile, 256 threads = 8 waves (4 row-waves x 2 col-waves),
// wave tile = 32x64 = 2x4 WMMA accumulators. K stepped by 32, double-buffered LDS.
// EPI: 0 = GELU -> bf16 ; 1 = identity -> fp32 + bf16 ; 2 = sigmoid -> fp32 + MSE loss
// ---------------------------------------------------------------------------
template <int EPI>
__global__ __launch_bounds__(256) void k_gemm(
    const bf16* __restrict__ A, const bf16* __restrict__ Bt,
    const float* __restrict__ bias,
    bf16* __restrict__ outB, float* __restrict__ outF,
    const float* __restrict__ xref, float* __restrict__ lossAcc,
    int M, int N, int K)
{
  __shared__ alignas(16) bf16 As[2][128 * 32];
  __shared__ alignas(16) bf16 Bs[2][128 * 32];
  __shared__ float lred[256];

  const int tid  = threadIdx.x;
  const int lane = tid & 31, warp = tid >> 5;
  const int wr = warp >> 1, wc = warp & 1;
  const int l15 = lane & 15, h = lane >> 4;
  const size_t rowBase = (size_t)blockIdx.y * 128;
  const size_t colBase = (size_t)blockIdx.x * 128;
  const int kTiles = K >> 5;

  const v8f vz = {0.f, 0.f, 0.f, 0.f, 0.f, 0.f, 0.f, 0.f};
  v8f acc[2][4];
  for (int i = 0; i < 2; ++i)
    for (int j = 0; j < 4; ++j) acc[i][j] = vz;

  auto compute = [&](int cur) {
    v16bf af[2], bfr[4];
    for (int tr = 0; tr < 2; ++tr)
      af[tr] = ld_frag(&As[cur][(wr * 32 + tr * 16 + l15) * 32], h);
    for (int tc = 0; tc < 4; ++tc)
      bfr[tc] = ld_frag(&Bs[cur][(wc * 64 + tc * 16 + l15) * 32], h);
    for (int tr = 0; tr < 2; ++tr)
      for (int tc = 0; tc < 4; ++tc)
        acc[tr][tc] = wmma_bf16(af[tr], bfr[tc], acc[tr][tc]);
  };

#if USE_TDM
  if (warp == 0) {
    tdm_load_2d(A + rowBase * K, lds_off(&As[0][0]), K, 128, 32);
    tdm_load_2d(Bt + colBase * K, lds_off(&Bs[0][0]), K, 128, 32);
  }
  for (int kt = 0; kt < kTiles; ++kt) {
    const int cur = kt & 1, nxt = cur ^ 1;
    if (warp == 0) {
      if (kt + 1 < kTiles) {
        tdm_load_2d(A + rowBase * K + (size_t)(kt + 1) * 32, lds_off(&As[nxt][0]), K, 128, 32);
        tdm_load_2d(Bt + colBase * K + (size_t)(kt + 1) * 32, lds_off(&Bs[nxt][0]), K, 128, 32);
        __builtin_amdgcn_s_wait_tensorcnt(2);   // current pair done (in-order)
      } else {
        __builtin_amdgcn_s_wait_tensorcnt(0);
      }
    }
    __syncthreads();
    compute(cur);
    __syncthreads();   // buffer 'nxt' free for the issue at the next iteration
  }
#else
  auto loadTiles = [&](int buf, int kt) {
    const int r = tid >> 1, cH = (tid & 1) * 16;
    const uint4* ga = (const uint4*)(A + (rowBase + r) * K + (size_t)kt * 32 + cH);
    uint4 a0 = ga[0], a1 = ga[1];
    const uint4* gb = (const uint4*)(Bt + (colBase + r) * K + (size_t)kt * 32 + cH);
    uint4 b0 = gb[0], b1 = gb[1];
    uint4* sa = (uint4*)&As[buf][r * 32 + cH]; sa[0] = a0; sa[1] = a1;
    uint4* sb = (uint4*)&Bs[buf][r * 32 + cH]; sb[0] = b0; sb[1] = b1;
  };
  loadTiles(0, 0);
  for (int kt = 0; kt < kTiles; ++kt) {
    __syncthreads();
    compute(kt & 1);
    if (kt + 1 < kTiles) loadTiles((kt + 1) & 1, kt + 1);
  }
#endif

  // Epilogue. D layout: VGPR r -> row = base + tr*16 + 8*h + r, col = base + tc*16 + l15.
  float lsum = 0.f;
  const size_t rw0 = rowBase + (size_t)wr * 32 + 8 * h;
  for (int tr = 0; tr < 2; ++tr) {
    for (int tc = 0; tc < 4; ++tc) {
      const int col = (int)colBase + wc * 64 + tc * 16 + l15;
      const float bv = bias[col];
      for (int r = 0; r < 8; ++r) {
        const size_t row = rw0 + tr * 16 + r;
        const float v = acc[tr][tc][r] + bv;
        if constexpr (EPI == 0) {
          outB[row * (size_t)N + col] = (bf16)gelu_exact(v);
        } else if constexpr (EPI == 1) {
          outF[row * (size_t)N + col] = v;
          outB[row * (size_t)N + col] = (bf16)v;
        } else {
          const float p = sigmoid_(v);
          outF[row * (size_t)N + col] = p;
          const float d = xref[row * (size_t)N + col] - p;
          lsum += d * d;
        }
      }
    }
  }
  if constexpr (EPI == 2) {
    lred[tid] = lsum;
    __syncthreads();
    for (int s = 128; s > 0; s >>= 1) {
      if (tid < s) lred[tid] += lred[tid + s];
      __syncthreads();
    }
    if (tid == 0) atomicAdd(lossAcc, lred[0]);
  }
}

// ---------------------------------------------------------------------------
// Fused codebook search: argmin_k ||z_e - e_k||^2 == argmin_k (||e_k||^2 - 2 z_e.e_k).
// Block = 64 rows of z_e (panel resident in LDS), loops over all 4096 codes in
// 128-wide chunks via WMMA; running best folded in registers, reduced via
// shfl_xor (16-lane half-groups) + LDS.  8 waves = 2 row-waves x 4 col-waves.
// ---------------------------------------------------------------------------
__global__ __launch_bounds__(256) void k_vq_argmin(
    const bf16* __restrict__ zeb, const bf16* __restrict__ eb,
    const float* __restrict__ enorm, int* __restrict__ zidx)
{
  constexpr int L = 256, KC = 4096;
  __shared__ alignas(16) bf16 Ap[64 * 256];        // 32 KB z_e panel
  __shared__ alignas(16) bf16 Bs[2][128 * 32];     // 16 KB double-buffered code tiles
  __shared__ float rs[64][4];
  __shared__ int   ri[64][4];

  const int tid  = threadIdx.x;
  const int lane = tid & 31, warp = tid >> 5;
  const int wr = warp >> 2, wc = warp & 3;
  const int l15 = lane & 15, h = lane >> 4;
  const size_t rowBase = (size_t)blockIdx.x * 64;

  float bsc[2][8];
  int   bix[2][8];
  for (int i = 0; i < 2; ++i)
    for (int r = 0; r < 8; ++r) { bsc[i][r] = 3.402823e38f; bix[i][r] = 0; }

  const v8f vz = {0.f, 0.f, 0.f, 0.f, 0.f, 0.f, 0.f, 0.f};
  v8f acc[2][2];
  const int S = (KC / 128) * (L / 32);  // 32 chunks * 8 k-steps = 256

  auto compute = [&](int cur, int kt) {
    v16bf af[2], bfr[2];
    for (int tr = 0; tr < 2; ++tr)
      af[tr] = ld_frag(&Ap[(wr * 32 + tr * 16 + l15) * 256 + kt * 32], h);
    for (int tc = 0; tc < 2; ++tc)
      bfr[tc] = ld_frag(&Bs[cur][(wc * 32 + tc * 16 + l15) * 32], h);
    for (int tr = 0; tr < 2; ++tr)
      for (int tc = 0; tc < 2; ++tc)
        acc[tr][tc] = wmma_bf16(af[tr], bfr[tc], acc[tr][tc]);
  };
  auto fold = [&](int c) {
    for (int tr = 0; tr < 2; ++tr)
      for (int tc = 0; tc < 2; ++tc) {
        const int col = c * 128 + wc * 32 + tc * 16 + l15;
        const float en = enorm[col];
        for (int r = 0; r < 8; ++r) {
          const float sd = en - 2.0f * acc[tr][tc][r];
          if (sd < bsc[tr][r]) { bsc[tr][r] = sd; bix[tr][r] = col; }
        }
      }
  };

#if USE_TDM
  if (warp == 0) {
    tdm_load_2d(zeb + rowBase * L, lds_off(&Ap[0]), L, 64, 256);
    tdm_load_2d(eb, lds_off(&Bs[0][0]), L, 128, 32);
  }
  for (int s = 0; s < S; ++s) {
    const int c = s >> 3, kt = s & 7, cur = s & 1;
    if (kt == 0)
      for (int i = 0; i < 2; ++i)
        for (int j = 0; j < 2; ++j) acc[i][j] = vz;
    if (warp == 0) {
      if (s + 1 < S) {
        const int c2 = (s + 1) >> 3, kt2 = (s + 1) & 7;
        tdm_load_2d(eb + ((size_t)c2 * 128) * L + (size_t)kt2 * 32,
                    lds_off(&Bs[cur ^ 1][0]), L, 128, 32);
        __builtin_amdgcn_s_wait_tensorcnt(1);
      } else {
        __builtin_amdgcn_s_wait_tensorcnt(0);
      }
    }
    __syncthreads();
    compute(cur, kt);
    __syncthreads();
    if (kt == 7) fold(c);
  }
#else
  {  // z_e panel: 64x256 elems, 256 threads x 8 x uint4
    const int r = tid >> 2, q = tid & 3;
    const uint4* g = (const uint4*)(zeb + (rowBase + r) * L + q * 64);
    uint4* sp = (uint4*)&Ap[r * 256 + q * 64];
    for (int i = 0; i < 8; ++i) sp[i] = g[i];
  }
  auto loadB = [&](int buf, int c, int kt) {
    const int r = tid >> 1, cH = (tid & 1) * 16;
    const uint4* g = (const uint4*)(eb + ((size_t)c * 128 + r) * L + (size_t)kt * 32 + cH);
    uint4 b0 = g[0], b1 = g[1];
    uint4* sp = (uint4*)&Bs[buf][r * 32 + cH];
    sp[0] = b0; sp[1] = b1;
  };
  loadB(0, 0, 0);
  for (int s = 0; s < S; ++s) {
    const int c = s >> 3, kt = s & 7;
    if (kt == 0)
      for (int i = 0; i < 2; ++i)
        for (int j = 0; j < 2; ++j) acc[i][j] = vz;
    __syncthreads();
    compute(s & 1, kt);
    if (s + 1 < S) loadB((s + 1) & 1, (s + 1) >> 3, (s + 1) & 7);
    if (kt == 7) fold(c);
  }
#endif

  // Reduce across the 16 lanes of each half (same rows), then across col-waves.
  for (int tr = 0; tr < 2; ++tr)
    for (int r = 0; r < 8; ++r) {
      float s = bsc[tr][r];
      int   ix = bix[tr][r];
      for (int m = 8; m >= 1; m >>= 1) {
        const float os = __shfl_xor(s, m, 32);
        const int   oi = __shfl_xor(ix, m, 32);
        if (os < s || (os == s && oi < ix)) { s = os; ix = oi; }
      }
      if (l15 == 0) {
        const int rloc = wr * 32 + tr * 16 + 8 * h + r;
        rs[rloc][wc] = s;
        ri[rloc][wc] = ix;
      }
    }
  __syncthreads();
  if (tid < 64) {
    float s = rs[tid][0];
    int   ix = ri[tid][0];
    for (int j = 1; j < 4; ++j) {
      const float os = rs[tid][j];
      const int   oi = ri[tid][j];
      if (os < s || (os == s && oi < ix)) { s = os; ix = oi; }
    }
    zidx[rowBase + tid] = ix;
  }
}

// ---------------------------------------------------------------------------
// Small utility kernels
// ---------------------------------------------------------------------------
__global__ void k_cvt_bf16(const float* __restrict__ in, bf16* __restrict__ out, long n) {
  long i = (long)blockIdx.x * blockDim.x + threadIdx.x;
  const long stride = (long)gridDim.x * blockDim.x;
  for (; i < n; i += stride) out[i] = (bf16)in[i];
}

// out[n][k] = (bf16) in[k][n]    (in: K x N row-major fp32)
__global__ void k_transpose_cvt(const float* __restrict__ in, bf16* __restrict__ out,
                                int K, int N) {
  __shared__ float t[32][33];
  const int kb = blockIdx.y * 32, nb = blockIdx.x * 32;
  const int tx = threadIdx.x, ty = threadIdx.y;  // 32 x 8
  for (int i = ty; i < 32; i += 8) t[i][tx] = in[(size_t)(kb + i) * N + nb + tx];
  __syncthreads();
  for (int i = ty; i < 32; i += 8) out[(size_t)(nb + i) * K + kb + tx] = (bf16)t[tx][i];
}

__global__ void k_enorm(const float* __restrict__ emb, float* __restrict__ en, int L) {
  const int code = blockIdx.x * 8 + (threadIdx.x >> 5);
  const int lane = threadIdx.x & 31;
  float s = 0.f;
  for (int l = lane; l < L; l += 32) {
    const float e = emb[(size_t)code * L + l];
    s += e * e;
  }
  for (int m = 16; m; m >>= 1) s += __shfl_xor(s, m, 32);
  if (lane == 0) en[code] = s;
}

__global__ void k_zero_i4(int4* __restrict__ p, long n4) {
  long i = (long)blockIdx.x * blockDim.x + threadIdx.x;
  const long stride = (long)gridDim.x * blockDim.x;
  const int4 z = {0, 0, 0, 0};
  for (; i < n4; i += stride) p[i] = z;
}

__global__ void k_onehot(const int* __restrict__ zidx, int* __restrict__ disc, int KC) {
  const int row = blockIdx.x * 256 + threadIdx.x;
  disc[(size_t)row * KC + zidx[row]] = 1;
}

// z_q gather (bf16 for decoder) + VQ loss: 1.25 * sum((z_e - z_q)^2)
__global__ __launch_bounds__(256) void k_zq_gather_loss(
    const int* __restrict__ zidx, const float* __restrict__ emb,
    const float* __restrict__ zef, bf16* __restrict__ zqb,
    float* __restrict__ lossAcc, int L)
{
  __shared__ float sred[256];
  const int row = blockIdx.x, t = threadIdx.x;
  const int idx = zidx[row];
  const float e = emb[(size_t)idx * L + t];
  zqb[(size_t)row * L + t] = (bf16)e;
  const float d = zef[(size_t)row * L + t] - e;
  sred[t] = 1.25f * d * d;
  __syncthreads();
  for (int s = 128; s > 0; s >>= 1) {
    if (t < s) sred[t] += sred[t + s];
    __syncthreads();
  }
  if (t == 0) atomicAdd(lossAcc, sred[0]);
}

__global__ void k_init_loss(float* acc) { *acc = 0.f; }
__global__ void k_final_loss(const float* acc, float* out, float n) { *out = *acc / n; }

// ---------------------------------------------------------------------------
// Host-side orchestration
// ---------------------------------------------------------------------------
extern "C" void kernel_launch(void* const* d_in, const int* in_sizes, int n_in,
                              void* d_out, int out_size, void* d_ws, size_t ws_size,
                              hipStream_t stream)
{
  (void)in_sizes; (void)n_in; (void)out_size; (void)ws_size;
  constexpr int NB = 16384, DD = 1024, HH = 1024, LL = 256, KC = 4096;

  const float* x   = (const float*)d_in[0];
  const float* emb = (const float*)d_in[1];
  const float* ew1 = (const float*)d_in[2];
  const float* eb1 = (const float*)d_in[3];
  const float* ew2 = (const float*)d_in[4];
  const float* eb2 = (const float*)d_in[5];
  const float* ew3 = (const float*)d_in[6];
  const float* eb3 = (const float*)d_in[7];
  const float* dw1 = (const float*)d_in[8];
  const float* db1 = (const float*)d_in[9];
  const float* dw2 = (const float*)d_in[10];
  const float* db2 = (const float*)d_in[11];
  const float* dw3 = (const float*)d_in[12];
  const float* db3 = (const float*)d_in[13];

  float* xpred = (float*)d_out;                                  // [N, D] fp32
  int*   disc  = (int*)(xpred + (size_t)NB * DD);                // [N, K] int32 one-hot
  float* lossO = ((float*)d_out) + (size_t)NB * DD + (size_t)NB * KC;  // scalar

  char* w = (char*)d_ws;
  auto take = [&](size_t b) -> char* {
    char* p = w;
    w += (b + 255) & ~(size_t)255;
    return p;
  };
  bf16*  bufA  = (bf16*)take((size_t)NB * HH * 2);   // activation ping
  bf16*  bufB  = (bf16*)take((size_t)NB * HH * 2);   // activation pong
  bf16*  zebuf = (bf16*)take((size_t)NB * LL * 2);   // z_e bf16
  float* zef   = (float*)take((size_t)NB * LL * 4);  // z_e fp32
  bf16*  zqb   = (bf16*)take((size_t)NB * LL * 2);   // z_q bf16
  bf16*  w1t   = (bf16*)take((size_t)DD * HH * 2);
  bf16*  w2t   = (bf16*)take((size_t)HH * HH * 2);
  bf16*  w3t   = (bf16*)take((size_t)HH * LL * 2);
  bf16*  v1t   = (bf16*)take((size_t)LL * HH * 2);
  bf16*  v2t   = (bf16*)take((size_t)HH * HH * 2);
  bf16*  v3t   = (bf16*)take((size_t)HH * DD * 2);
  bf16*  embB  = (bf16*)take((size_t)KC * LL * 2);
  float* enorm = (float*)take((size_t)KC * 4);
  int*   zidx  = (int*)take((size_t)NB * 4);
  float* lacc  = (float*)take(256);

  // Init + precision staging (redone every call -> deterministic, graph-safe)
  k_init_loss<<<1, 1, 0, stream>>>(lacc);
  k_zero_i4<<<4096, 256, 0, stream>>>((int4*)disc, (long)NB * KC / 4);
  k_cvt_bf16<<<4096, 256, 0, stream>>>(x, bufA, (long)NB * DD);
  k_cvt_bf16<<<1024, 256, 0, stream>>>(emb, embB, (long)KC * LL);
  k_transpose_cvt<<<dim3(HH / 32, DD / 32), dim3(32, 8), 0, stream>>>(ew1, w1t, DD, HH);
  k_transpose_cvt<<<dim3(HH / 32, HH / 32), dim3(32, 8), 0, stream>>>(ew2, w2t, HH, HH);
  k_transpose_cvt<<<dim3(LL / 32, HH / 32), dim3(32, 8), 0, stream>>>(ew3, w3t, HH, LL);
  k_transpose_cvt<<<dim3(HH / 32, LL / 32), dim3(32, 8), 0, stream>>>(dw1, v1t, LL, HH);
  k_transpose_cvt<<<dim3(HH / 32, HH / 32), dim3(32, 8), 0, stream>>>(dw2, v2t, HH, HH);
  k_transpose_cvt<<<dim3(DD / 32, HH / 32), dim3(32, 8), 0, stream>>>(dw3, v3t, HH, DD);
  k_enorm<<<KC / 8, 256, 0, stream>>>(emb, enorm, LL);

  // Encoder
  k_gemm<0><<<dim3(HH / 128, NB / 128), 256, 0, stream>>>(bufA, w1t, eb1, bufB, nullptr, nullptr, nullptr, NB, HH, DD);
  k_gemm<0><<<dim3(HH / 128, NB / 128), 256, 0, stream>>>(bufB, w2t, eb2, bufA, nullptr, nullptr, nullptr, NB, HH, HH);
  k_gemm<1><<<dim3(LL / 128, NB / 128), 256, 0, stream>>>(bufA, w3t, eb3, zebuf, zef, nullptr, nullptr, NB, LL, HH);

  // Vector quantization (fused WMMA distance + argmin)
  k_vq_argmin<<<NB / 64, 256, 0, stream>>>(zebuf, embB, enorm, zidx);
  k_onehot<<<NB / 256, 256, 0, stream>>>(zidx, disc, KC);
  k_zq_gather_loss<<<NB, 256, 0, stream>>>(zidx, emb, zef, zqb, lacc, LL);

  // Decoder (straight-through: forward value of z_ss == z_q)
  k_gemm<0><<<dim3(HH / 128, NB / 128), 256, 0, stream>>>(zqb, v1t, db1, bufB, nullptr, nullptr, nullptr, NB, HH, LL);
  k_gemm<0><<<dim3(HH / 128, NB / 128), 256, 0, stream>>>(bufB, v2t, db2, bufA, nullptr, nullptr, nullptr, NB, HH, HH);
  k_gemm<2><<<dim3(DD / 128, NB / 128), 256, 0, stream>>>(bufA, v3t, db3, nullptr, xpred, x, lacc, NB, DD, HH);

  k_final_loss<<<1, 1, 0, stream>>>(lacc, lossO, (float)NB);
}
